// TorchFP4Linear_74088185856672
// MI455X (gfx1250) — matrix operand order repaired and checked by reading, not compiled, not run on hardware
//
#include <hip/hip_runtime.h>

typedef __attribute__((ext_vector_type(16))) _Float16 v16h;
typedef __attribute__((ext_vector_type(8)))  _Float16 v8h;
typedef __attribute__((ext_vector_type(2)))  _Float16 h2;
typedef __attribute__((ext_vector_type(8)))  float    v8f;

#define BM 128
#define BN 256
#define BK 64
#define LDT (BK + 8)   // LDS row stride in halves

// FP4 E2M1 -> f16 bit pattern. e==0: subnormal (0 or 0.5); else (2+m)*2^(e-2).
__device__ __forceinline__ unsigned fp4_half_bits(int c) {
    int e = (c >> 1) & 3, m = c & 1;
    unsigned mag = e ? (((unsigned)(e + 14) << 10) | ((unsigned)m << 9))
                     : (m ? 0x3800u : 0u);
    return ((unsigned)(c & 8) << 12) | mag;   // sign bit3 -> bit15
}
// One packed byte -> two f16 (even element = hi nibble in low half).
__device__ __forceinline__ unsigned fp4_pair_bits(int b) {
    return fp4_half_bits((b >> 4) & 15) | (fp4_half_bits(b & 15) << 16);
}

// A fragment (16x32 f16): lane L holds row M = L&15;
// halves 0..7 = K base..base+7, halves 8..15 = K base+16..base+23, base=8 for hi lanes.
__device__ __forceinline__ v16h load_frag_a(const _Float16* row, int lane, int kb) {
    int off = (lane & 16) >> 1;
    const v8h lo = *(const v8h*)(row + kb + off);
    const v8h hi = *(const v8h*)(row + kb + off + 16);
    v16h r;
#pragma unroll
    for (int i = 0; i < 8; ++i) { r[i] = lo[i]; r[i + 8] = hi[i]; }
    return r;
}

// B fragment (32x16 f16): lane L holds column N = L&15;
// halves 0..7 = K base..base+7, halves 8..15 = K base+16..base+23, base=8 for hi lanes.
__device__ __forceinline__ v16h load_frag_b(const _Float16* row, int lane, int kb) {
    int off = (lane & 16) >> 1;
    const v8h lo = *(const v8h*)(row + kb + off);
    const v8h hi = *(const v8h*)(row + kb + off + 16);
    v16h r;
#pragma unroll
    for (int i = 0; i < 8; ++i) { r[i] = lo[i]; r[i + 8] = hi[i]; }
    return r;
}

__global__ __launch_bounds__(256)
void fp4_linear_wmma(const float* __restrict__ x,
                     const int*   __restrict__ wq,
                     const float* __restrict__ wscale,
                     const float* __restrict__ bias,
                     float*       __restrict__ out,
                     int T, int O, int I)
{
    __shared__ _Float16 xs [BM * LDT];   // 18 KB
    __shared__ _Float16 wsh[BN * LDT];   // 36 KB

    const int t    = threadIdx.x;
    const int lane = t & 31;
    const int wave = t >> 5;
    const int wm   = wave & 1;           // 0..1 -> M offset 64*wm
    const int wn   = wave >> 1;          // 0..3 -> N offset 64*wn
    const int lr   = lane & 15;

    const int m0 = blockIdx.y * BM;
    const int n0 = blockIdx.x * BN;
    const int I2 = I >> 1;
    const int IB = I >> 4;

    // ---- per-thread staging geometry ----
    // x: thread owns 8 consecutive floats at col (t&7)*8 in rows (t>>3)+32i, i=0..3
    const float* xbase = x + (size_t)(m0 + (t >> 3)) * I + ((t & 7) << 3);
    // w: thread owns the full 64-wide K row of output row n0+t (32 packed ints)
    const int*   wbase = wq + (size_t)(n0 + t) * I2;
    const float* sbase = wscale + (size_t)(n0 + t) * IB;

    float4 xr[8];   // 4 segments x 2 float4
    int4   wr[8];   // 32 packed ints
    float4 sv;      // 4 block scales

    auto issue_loads = [&](int k0) {
#pragma unroll
        for (int i = 0; i < 4; ++i) {
            const float* p = xbase + (size_t)(32 * i) * I + k0;
            xr[2 * i + 0] = *(const float4*)(p);
            xr[2 * i + 1] = *(const float4*)(p + 4);
        }
#pragma unroll
        for (int i = 0; i < 8; ++i)
            wr[i] = *(const int4*)(wbase + (k0 >> 1) + 4 * i);
        sv = *(const float4*)(sbase + (k0 >> 4));
    };

    auto stage_store = [&]() {
        // x tile: one ds_store_b128 per row segment
#pragma unroll
        for (int i = 0; i < 4; ++i) {
            v8h p;
            p[0] = (_Float16)xr[2 * i].x;     p[1] = (_Float16)xr[2 * i].y;
            p[2] = (_Float16)xr[2 * i].z;     p[3] = (_Float16)xr[2 * i].w;
            p[4] = (_Float16)xr[2 * i + 1].x; p[5] = (_Float16)xr[2 * i + 1].y;
            p[6] = (_Float16)xr[2 * i + 1].z; p[7] = (_Float16)xr[2 * i + 1].w;
            *(v8h*)(&xs[((t >> 3) + 32 * i) * LDT + ((t & 7) << 3)]) = p;
        }
        // weight tile: dequant 32 bytes -> 64 f16 (32 contiguous dwords)
        h2 sc[4];
#pragma unroll
        for (int b = 0; b < 4; ++b) {
            _Float16 s = (_Float16)((const float*)&sv)[b];
            sc[b][0] = s; sc[b][1] = s;
        }
        unsigned* wd = (unsigned*)&wsh[t * LDT];
        const int* wb = (const int*)wr;   // 32 packed bytes (one per int)
#pragma unroll
        for (int j = 0; j < 32; ++j) {
            h2 v = __builtin_bit_cast(h2, fp4_pair_bits(wb[j]));
            v = v * sc[j >> 3];                      // v_pk_mul_f16
            wd[j] = __builtin_bit_cast(unsigned, v);
        }
    };

    v8f acc[4][4];
#pragma unroll
    for (int i = 0; i < 4; ++i)
#pragma unroll
        for (int j = 0; j < 4; ++j) {
            v8f z;
#pragma unroll
            for (int e = 0; e < 8; ++e) z[e] = 0.0f;
            acc[i][j] = z;
        }

    issue_loads(0);                      // prologue: stage 0 in flight

    for (int k0 = 0; k0 < I; k0 += BK) {
        stage_store();                   // consume regs (latency already hidden)
        __syncthreads();

        if (k0 + BK < I) issue_loads(k0 + BK);   // next stage in flight over compute

#pragma unroll
        for (int kb = 0; kb < BK; kb += 32) {
            v16h A[4];
#pragma unroll
            for (int mf = 0; mf < 4; ++mf)
                A[mf] = load_frag_a(&xs[(wm * 64 + mf * 16 + lr) * LDT], lane, kb);
#pragma unroll
            for (int nf = 0; nf < 4; ++nf) {
                v16h b = load_frag_b(&wsh[(wn * 64 + nf * 16 + lr) * LDT], lane, kb);
#pragma unroll
                for (int mf = 0; mf < 4; ++mf) {
                    acc[mf][nf] = __builtin_amdgcn_wmma_f32_16x16x32_f16(
                        false, A[mf], false, b, (short)0, acc[mf][nf], false, false);
                }
            }
        }
        __syncthreads();
    }

    // ---- epilogue: C/D layout lane = N (mod 16), VGPR r = M row (+8 for hi lanes) ----
    const int lhi = (lane >> 4) << 3;
#pragma unroll
    for (int mf = 0; mf < 4; ++mf) {
#pragma unroll
        for (int nf = 0; nf < 4; ++nf) {
            int n = n0 + wn * 64 + nf * 16 + lr;
            float bv = bias[n];
            size_t base = (size_t)(m0 + wm * 64 + mf * 16 + lhi) * O + n;
#pragma unroll
            for (int r = 0; r < 8; ++r) {
                out[base + (size_t)r * O] = acc[mf][nf][r] + bv;
            }
        }
    }
}

extern "C" void kernel_launch(void* const* d_in, const int* in_sizes, int n_in,
                              void* d_out, int out_size, void* d_ws, size_t ws_size,
                              hipStream_t stream) {
    const float* x      = (const float*)d_in[0];
    const int*   wq     = (const int*)d_in[1];
    const float* wscale = (const float*)d_in[2];
    const float* bias   = (const float*)d_in[3];
    float* out = (float*)d_out;

    const int O = in_sizes[3];                               // 4096
    const int I = (int)((2LL * (long long)in_sizes[1]) / O); // 4096
    const int T = (int)((long long)in_sizes[0] / I);         // 8192

    dim3 grid(O / BN, T / BM);   // (16, 64)
    dim3 block(256);
    fp4_linear_wmma<<<grid, block, 0, stream>>>(x, wq, wscale, bias, out, T, O, I);
}
